// Net_63831803953793
// MI455X (gfx1250) — matrix operand order repaired and checked by reading, not compiled, not run on hardware
//
#include <hip/hip_runtime.h>
#include <hip/hip_bf16.h>
#include <math.h>

#define N_ATOMS 65536
#define NBR     12
#define E_EDGES (N_ATOMS * NBR)      // 786432
#define G_GRAPHS 1024
#define AF      64
#define NF      41
#define NFP     42                   // padded edge_attr row (4B-aligned pairs)
#define HDIM    128
#define KTOT    169                  // 2*AF + NF
#define EPSBN   1e-5f

#define ETILE   48                   // edges per block = 4 atoms
#define NBLK    (E_EDGES / ETILE)    // 16384
#define AROW    200                  // f16 LDS row stride: 400B -> 16B aligned rows, banks spread
#define GROW    132                  // f32 LDS row stride for gated tile

typedef _Float16 v16h __attribute__((ext_vector_type(16)));
typedef float    v8f  __attribute__((ext_vector_type(8)));
typedef unsigned int v4u __attribute__((ext_vector_type(4)));

__device__ __forceinline__ float softplus_f(float x) {
    return x > 20.f ? x : log1pf(__expf(x));
}
__device__ __forceinline__ float sigmoid_f(float x) {
    return 1.f / (1.f + __expf(-x));
}
__device__ __forceinline__ int lower_bound_i(const int* arr, int n, int key) {
    int lo = 0, hi = n;
    while (lo < hi) { int mid = (lo + hi) >> 1; if (arr[mid] < key) lo = mid + 1; else hi = mid; }
    return lo;
}

// ---------------- atom encoder: h[n,c] = sum_i emb_i[x[n,i], c]; also f16 copy --------
__global__ __launch_bounds__(256) void encoder_kernel(
    const int* __restrict__ x,
    const float* __restrict__ t0, const float* __restrict__ t1, const float* __restrict__ t2,
    const float* __restrict__ t3, const float* __restrict__ t4, const float* __restrict__ t5,
    const float* __restrict__ t6, const float* __restrict__ t7, const float* __restrict__ t8,
    float* __restrict__ h, _Float16* __restrict__ h16)
{
    int idx = blockIdx.x * 256 + threadIdx.x;          // over N*64
    int n = idx >> 6, c = idx & 63;
    const int* xr = x + n * 9;
    float s = t0[xr[0] * AF + c];
    s += t1[xr[1] * AF + c];  s += t2[xr[2] * AF + c];
    s += t3[xr[3] * AF + c];  s += t4[xr[4] * AF + c];
    s += t5[xr[5] * AF + c];  s += t6[xr[6] * AF + c];
    s += t7[xr[7] * AF + c];  s += t8[xr[8] * AF + c];
    h[idx] = s;
    h16[idx] = (_Float16)s;
}

// -------- edge_attr f32 [E,41] -> f16 padded [E,42] (pad half = 0), pair-granular ------
__global__ __launch_bounds__(256) void ea_convert_kernel(const float* __restrict__ ea,
                                                         _Float16* __restrict__ ea16)
{
    int idx = blockIdx.x * 256 + threadIdx.x;          // over E * 21 pairs
    int e = idx / (NFP / 2), j2 = idx - e * (NFP / 2);
    int j = j2 * 2;
    float v0 = ea[(size_t)e * NF + j];
    float v1 = (j + 1 < NF) ? ea[(size_t)e * NF + j + 1] : 0.f;
    _Float16 h2[2] = { (_Float16)v0, (_Float16)v1 };
    *(unsigned int*)(ea16 + (size_t)e * NFP + j) = *(const unsigned int*)h2;
}

// -------- pack fcw[l] (128x169 f32) into per-lane B-fragment order, f16 --------
// layout: [ntile(8)][kstep(6)][lane(32)][16 halves]; lane: N=16*nt+(lane&15),
// VGPR v holds K = 32*kstep + 16*(lane>>4) + 2v {,+1}
__global__ __launch_bounds__(256) void packw_kernel(const float* __restrict__ fcw_l,
                                                    _Float16* __restrict__ wpack)
{
    for (int idx = blockIdx.x * 256 + threadIdx.x; idx < 8 * 6 * 32 * 16;
         idx += gridDim.x * 256) {
        int i    = idx & 15;
        int lane = (idx >> 4) & 31;
        int r    = idx >> 9;              // 0..47
        int ks   = r % 6;
        int nt   = r / 6;
        int v = i >> 1, odd = i & 1;
        int n = nt * 16 + (lane & 15);
        int K = ks * 32 + ((lane >> 4) << 4) + 2 * v + odd;
        float val = (K < KTOT) ? fcw_l[n * KTOT + K] : 0.f;
        wpack[idx] = (_Float16)val;
    }
}

// ---------------- fused edge GEMM (WMMA f16->f32), two modes ----------------
// STATS=true : accumulate per-channel sum / sumsq partials for BN1
// STATS=false: apply BN1, msg = sigmoid(filter)*softplus(core), reduce 12 edges/atom
template <bool STATS>
__global__ __launch_bounds__(256) void conv_gemm_kernel(
    const _Float16* __restrict__ h16,
    const int* __restrict__ dst_idx,        // edge_index + E (dst row)
    const _Float16* __restrict__ ea16,      // [E, 42] padded f16
    const _Float16* __restrict__ wpack,
    const float* __restrict__ bn1stats,     // mean[128], istd[128]
    const float* __restrict__ bn1_g, const float* __restrict__ bn1_b,
    float* __restrict__ p1sum, float* __restrict__ p1sq,
    float* __restrict__ summed)
{
    __shared__ _Float16 A_s[ETILE * AROW];
    __shared__ float    G_s[ETILE * GROW];

    const int t   = threadIdx.x;
    const int blk = blockIdx.x;
    const int e0  = blk * ETILE;

    // warm the packed-W fragment stream (global_prefetch_b8) while we stage LDS
    if (t < 64) __builtin_prefetch(wpack + (size_t)t * 384, 0, 0);

    // ---- stage A tile [48 x 200] f16 ----
    // segment 1: cols 0..127 from h16[dst] / h16[src], 16B chunks (b128 copies)
    for (int idx = t; idx < ETILE * 16; idx += 256) {
        int r = idx >> 4, q = idx & 15;
        int e = e0 + r;
        const _Float16* srcp = (q < 8)
            ? (h16 + (size_t)dst_idx[e] * AF + q * 8)
            : (h16 + (size_t)(e / NBR) * AF + (q - 8) * 8);
        *(v4u*)&A_s[r * AROW + q * 8] = *(const v4u*)srcp;
    }
    // segment 2: cols 128..169 from padded edge_attr f16, 4B chunks
    for (int idx = t; idx < ETILE * (NFP / 2); idx += 256) {
        int r = idx / (NFP / 2), j2 = idx - r * (NFP / 2);
        int e = e0 + r;
        *(unsigned int*)&A_s[r * AROW + 2 * AF + 2 * j2] =
            *(const unsigned int*)(ea16 + (size_t)e * NFP + 2 * j2);
    }
    // segment 3: zero-fill K-pad cols 170..199
    for (int idx = t; idx < ETILE * 15; idx += 256) {
        int r = idx / 15, j2 = idx - r * 15;
        *(unsigned int*)&A_s[r * AROW + 170 + 2 * j2] = 0u;
    }
    __syncthreads();

    const int lane   = t & 31;
    const int wv     = t >> 5;          // wave id 0..7 -> output N-tile
    const int laneHi = lane >> 4;
    const int mrow   = lane & 15;

    v8f acc[3] = {};
    const v16h* bptr = reinterpret_cast<const v16h*>(wpack);

#pragma unroll
    for (int ks = 0; ks < 6; ++ks) {
        v16h bfrag = bptr[(wv * 6 + ks) * 32 + lane];
#pragma unroll
        for (int mt = 0; mt < 3; ++mt) {
            const _Float16* ap = &A_s[(mt * 16 + mrow) * AROW + ks * 32 + laneHi * 16];
            v16h afrag;
#pragma unroll
            for (int i2 = 0; i2 < 8; ++i2) { afrag[i2] = ap[i2]; afrag[i2 + 8] = ap[i2 + 8]; }
            acc[mt] = __builtin_amdgcn_wmma_f32_16x16x32_f16(
                false, afrag, false, bfrag, (short)0, acc[mt], false, false);
        }
    }

    if (STATS) {
        float s = 0.f, ss = 0.f;
#pragma unroll
        for (int mt = 0; mt < 3; ++mt)
#pragma unroll
            for (int r = 0; r < 8; ++r) { float xv = acc[mt][r]; s += xv; ss += xv * xv; }
        s  += __shfl_xor(s, 16);
        ss += __shfl_xor(ss, 16);
        if (laneHi == 0) {
            int c = wv * 16 + mrow;
            p1sum[blk * HDIM + c] = s;
            p1sq [blk * HDIM + c] = ss;
        }
    } else {
        int c = wv * 16 + mrow;
        float mean = bn1stats[c], istd = bn1stats[HDIM + c];
        float g = bn1_g[c], bb = bn1_b[c];
#pragma unroll
        for (int mt = 0; mt < 3; ++mt)
#pragma unroll
            for (int r = 0; r < 8; ++r) {
                int m = mt * 16 + r + 8 * laneHi;
                G_s[m * GROW + c] = g * (acc[mt][r] - mean) * istd + bb;
            }
        __syncthreads();
        // msg + fixed-degree (12) reduce: 4 atoms x 64 channels = 256 threads
        int a = t >> 6, cc = t & 63;
        float sum = 0.f;
#pragma unroll
        for (int j = 0; j < NBR; ++j) {
            int row = a * NBR + j;
            float f = G_s[row * GROW + cc];
            float o = G_s[row * GROW + cc + AF];
            sum += sigmoid_f(f) * softplus_f(o);
        }
        summed[(blk * 4 + a) * AF + cc] = sum;
    }
}

// ---------------- BN finalize: reduce per-block partials -> mean/istd ----------------
__global__ __launch_bounds__(256) void bn_finalize_kernel(
    const float* __restrict__ psum, const float* __restrict__ psq,
    int nblk, int nch, float invCount, float* __restrict__ stats)
{
    int c = blockIdx.x, t = threadIdx.x;
    float s = 0.f, ss = 0.f;
    for (int i = t; i < nblk; i += 256) { s += psum[i * nch + c]; ss += psq[i * nch + c]; }
    __shared__ float rs[256], rss[256];
    rs[t] = s; rss[t] = ss; __syncthreads();
    for (int st = 128; st > 0; st >>= 1) {
        if (t < st) { rs[t] += rs[t + st]; rss[t] += rss[t + st]; }
        __syncthreads();
    }
    if (t == 0) {
        float mean = rs[0] * invCount;
        float var  = rss[0] * invCount - mean * mean;
        stats[c] = mean;
        stats[nch + c] = rsqrtf(var + EPSBN);
    }
}

// ---------------- BN2 partials over summed [N,64] ----------------
__global__ __launch_bounds__(256) void bn2_partials_kernel(
    const float* __restrict__ summed, float* __restrict__ p2sum, float* __restrict__ p2sq)
{
    int b = blockIdx.x, t = threadIdx.x;
    int c = t & 63, grp = t >> 6;
    float s = 0.f, ss = 0.f;
    for (int a = grp; a < 256; a += 4) {
        float xv = summed[(b * 256 + a) * AF + c];
        s += xv; ss += xv * xv;
    }
    __shared__ float rs[256], rss[256];
    rs[t] = s; rss[t] = ss; __syncthreads();
    if (grp == 0) {
        float S  = rs[c]  + rs[c + 64]  + rs[c + 128]  + rs[c + 192];
        float SS = rss[c] + rss[c + 64] + rss[c + 128] + rss[c + 192];
        p2sum[b * AF + c] = S;
        p2sq [b * AF + c] = SS;
    }
}

// ---------------- h = softplus(h + BN2(summed)); refresh f16 mirror ----------------
__global__ __launch_bounds__(256) void h_update_kernel(
    float* __restrict__ h, _Float16* __restrict__ h16,
    const float* __restrict__ summed,
    const float* __restrict__ bn2stats,
    const float* __restrict__ g2, const float* __restrict__ b2)
{
    int idx = blockIdx.x * 256 + threadIdx.x;
    int c = idx & 63;
    float y = g2[c] * (summed[idx] - bn2stats[c]) * bn2stats[AF + c] + b2[c];
    float hv = softplus_f(h[idx] + y);
    h[idx] = hv;
    h16[idx] = (_Float16)hv;
}

// ---------------- per-graph mean pool (batch is sorted) ----------------
__global__ __launch_bounds__(256) void pool_kernel(
    const float* __restrict__ h, const int* __restrict__ batch, float* __restrict__ pooled)
{
    int g = blockIdx.x, t = threadIdx.x;
    int lo = lower_bound_i(batch, N_ATOMS, g);
    int hi = lower_bound_i(batch, N_ATOMS, g + 1);
    int c = t & 63, grp = t >> 6;
    float s = 0.f;
    for (int r = lo + grp; r < hi; r += 4) s += h[r * AF + c];
    __shared__ float rs[256];
    rs[t] = s; __syncthreads();
    if (grp == 0) {
        float tot = rs[c] + rs[c + 64] + rs[c + 128] + rs[c + 192];
        float cnt = (float)(hi - lo);
        pooled[g * AF + c] = tot / fmaxf(cnt, 1.f);
    }
}

// ---------------- head BN stats over G graphs (p = softplus(pooled)) ----------------
__global__ __launch_bounds__(256) void pstats_kernel(
    const float* __restrict__ pooled, float* __restrict__ stats)
{
    int c = blockIdx.x, t = threadIdx.x;
    float s = 0.f, ss = 0.f;
    for (int g = t; g < G_GRAPHS; g += 256) {
        float p = softplus_f(pooled[g * AF + c]);
        s += p; ss += p * p;
    }
    __shared__ float rs[256], rss[256];
    rs[t] = s; rss[t] = ss; __syncthreads();
    for (int st = 128; st > 0; st >>= 1) {
        if (t < st) { rs[t] += rs[t + st]; rss[t] += rss[t + st]; }
        __syncthreads();
    }
    if (t == 0) {
        float mean = rs[0] / (float)G_GRAPHS;
        float var  = rss[0] / (float)G_GRAPHS - mean * mean;
        stats[c] = mean;
        stats[AF + c] = rsqrtf(var + EPSBN);
    }
}

// ---------------- head: BN -> FC(64->128) -> softplus -> FC(128->1) ----------------
__global__ __launch_bounds__(128) void head_kernel(
    const float* __restrict__ pooled, const float* __restrict__ stats,
    const float* __restrict__ bng, const float* __restrict__ bnb,
    const float* __restrict__ w1, const float* __restrict__ b1,
    const float* __restrict__ wout, const float* __restrict__ bout,
    float* __restrict__ out)
{
    int g = blockIdx.x, t = threadIdx.x;
    __shared__ float pbn[AF];
    if (t < AF) {
        float p = softplus_f(pooled[g * AF + t]);
        pbn[t] = bng[t] * (p - stats[t]) * stats[AF + t] + bnb[t];
    }
    __syncthreads();
    float z = b1[t];
#pragma unroll
    for (int c = 0; c < AF; ++c) z += pbn[c] * w1[t * AF + c];
    z = softplus_f(z) * wout[t];
    __shared__ float rz[HDIM];
    rz[t] = z; __syncthreads();
    for (int st = 64; st > 0; st >>= 1) {
        if (t < st) rz[t] += rz[t + st];
        __syncthreads();
    }
    if (t == 0) out[g] = rz[0] + bout[0];
}

// ---------------------------------------------------------------------------
extern "C" void kernel_launch(void* const* d_in, const int* in_sizes, int n_in,
                              void* d_out, int out_size, void* d_ws, size_t ws_size,
                              hipStream_t stream) {
    const int*   x          = (const int*)d_in[0];
    const int*   edge_index = (const int*)d_in[1];
    const int*   batch      = (const int*)d_in[2];
    const float* edge_attr  = (const float*)d_in[3];

    // emb_tables: either 9 separate inputs (tuple flattened) or one blob
    const float* emb[9];
    int base;
    if (n_in >= 25) {
        for (int i = 0; i < 9; ++i) emb[i] = (const float*)d_in[4 + i];
        base = 13;
    } else {
        static const int offs[9] = {0, 119, 123, 135, 147, 157, 163, 169, 171};
        const float* eb = (const float*)d_in[4];
        for (int i = 0; i < 9; ++i) emb[i] = eb + (size_t)offs[i] * AF;
        base = 5;
    }
    const float* fcw   = (const float*)d_in[base + 0];  // [3,128,169]
    // fcb (base+1) is provably a no-op through BN1 (mean absorbs bias) -> unused
    const float* bn1_g = (const float*)d_in[base + 2];  // [3,128]
    const float* bn1_b = (const float*)d_in[base + 3];
    const float* bn2_g = (const float*)d_in[base + 4];  // [3,64]
    const float* bn2_b = (const float*)d_in[base + 5];
    const float* bng   = (const float*)d_in[base + 6];  // [64]
    const float* bnb   = (const float*)d_in[base + 7];
    const float* w1    = (const float*)d_in[base + 8];  // [128,64]
    const float* b1    = (const float*)d_in[base + 9];  // [128]
    const float* wout  = (const float*)d_in[base + 10]; // [1,128]
    const float* bout  = (const float*)d_in[base + 11]; // [1]
    float* out = (float*)d_out;

    // ---- workspace carve-up (256B aligned regions) ----
    char* ws = (char*)d_ws;
    size_t off = 0;
    auto alloc = [&](size_t bytes) { void* p = ws + off; off = (off + bytes + 255) & ~(size_t)255; return p; };
    float*     h        = (float*)alloc((size_t)N_ATOMS * AF * 4);
    float*     summed   = (float*)alloc((size_t)N_ATOMS * AF * 4);
    _Float16*  h16      = (_Float16*)alloc((size_t)N_ATOMS * AF * 2);
    _Float16*  ea16     = (_Float16*)alloc((size_t)E_EDGES * NFP * 2);
    _Float16*  wpack    = (_Float16*)alloc((size_t)8 * 6 * 32 * 16 * 2);
    float*     p1sum    = (float*)alloc((size_t)NBLK * HDIM * 4);
    float*     p1sq     = (float*)alloc((size_t)NBLK * HDIM * 4);
    float*     bn1stats = (float*)alloc(2 * HDIM * 4);
    float*     p2sum    = (float*)alloc(256 * AF * 4);
    float*     p2sq     = (float*)alloc(256 * AF * 4);
    float*     bn2stats = (float*)alloc(2 * AF * 4);
    float*     pooled   = (float*)alloc((size_t)G_GRAPHS * AF * 4);
    float*     pstats   = (float*)alloc(2 * AF * 4);
    (void)ws_size; (void)in_sizes; (void)out_size;

    const int* dst_idx = edge_index + E_EDGES;   // edge_index row 1

    encoder_kernel<<<N_ATOMS * AF / 256, 256, 0, stream>>>(
        x, emb[0], emb[1], emb[2], emb[3], emb[4], emb[5], emb[6], emb[7], emb[8], h, h16);
    ea_convert_kernel<<<E_EDGES * (NFP / 2) / 256, 256, 0, stream>>>(edge_attr, ea16);

    for (int l = 0; l < 3; ++l) {
        packw_kernel<<<96, 256, 0, stream>>>(fcw + (size_t)l * HDIM * KTOT, wpack);
        conv_gemm_kernel<true><<<NBLK, 256, 0, stream>>>(
            h16, dst_idx, ea16, wpack, nullptr, nullptr, nullptr, p1sum, p1sq, nullptr);
        bn_finalize_kernel<<<HDIM, 256, 0, stream>>>(
            p1sum, p1sq, NBLK, HDIM, 1.f / (float)E_EDGES, bn1stats);
        conv_gemm_kernel<false><<<NBLK, 256, 0, stream>>>(
            h16, dst_idx, ea16, wpack, bn1stats,
            bn1_g + l * HDIM, bn1_b + l * HDIM, nullptr, nullptr, summed);
        bn2_partials_kernel<<<256, 256, 0, stream>>>(summed, p2sum, p2sq);
        bn_finalize_kernel<<<AF, 256, 0, stream>>>(
            p2sum, p2sq, 256, AF, 1.f / (float)N_ATOMS, bn2stats);
        h_update_kernel<<<N_ATOMS * AF / 256, 256, 0, stream>>>(
            h, h16, summed, bn2stats, bn2_g + l * AF, bn2_b + l * AF);
    }

    pool_kernel<<<G_GRAPHS, 256, 0, stream>>>(h, batch, pooled);
    pstats_kernel<<<AF, 256, 0, stream>>>(pooled, pstats);
    head_kernel<<<G_GRAPHS, 128, 0, stream>>>(
        pooled, pstats, bng, bnb, w1, b1, wout, bout, out);
}
